// SkeletonTransformer_22995254903058
// MI455X (gfx1250) — matrix operand order repaired and checked by reading, not compiled, not run on hardware
//
#include <hip/hip_runtime.h>
#include <hip/hip_bf16.h>
#include <cstdint>

// ---------------------------------------------------------------------------
// Model constants
// ---------------------------------------------------------------------------
namespace {
constexpr int SEQ   = 25;
constexpr int BATCH = 512;
constexpr int DM    = 1024;
constexpr int NH    = 16;
constexpr int HD    = 64;
constexpr int NL    = 6;
constexpr int INAG  = 50;
constexpr int SB    = SEQ * BATCH;            // 12800 rows
constexpr size_t SBD = (size_t)SB * DM;       // 13,107,200 elems

constexpr int TM = 128, TN = 128, TK = 32;    // GEMM tile
constexpr int NK = DM / TK;                   // 32 K-steps
}

typedef __attribute__((ext_vector_type(16))) _Float16 v16h;
typedef __attribute__((ext_vector_type(8)))  _Float16 h8;   // 16 bytes
typedef __attribute__((ext_vector_type(8)))  float    v8f;
typedef __attribute__((ext_vector_type(4)))  float    f4;
typedef __attribute__((ext_vector_type(4)))  int      i4;

union F16Frag { v16h v; i4 q[2]; };

static __device__ __forceinline__ h8 pack8(f4 a, f4 b) {
  h8 r;
  r[0] = (_Float16)a[0]; r[1] = (_Float16)a[1];
  r[2] = (_Float16)a[2]; r[3] = (_Float16)a[3];
  r[4] = (_Float16)b[0]; r[5] = (_Float16)b[1];
  r[6] = (_Float16)b[2]; r[7] = (_Float16)b[3];
  return r;
}

// ---------------------------------------------------------------------------
// Weight prep: f32 [l][k][n] -> f16 transposed [l][n][k] (tiled via LDS)
// grid = (D/32, D/32, NL), block = (32, 8)
// ---------------------------------------------------------------------------
__global__ __launch_bounds__(256) void convert_transpose_kernel(
    const float* __restrict__ in, _Float16* __restrict__ out) {
  __shared__ _Float16 tile[32][33];
  int l  = blockIdx.z;
  int kb = blockIdx.x * 32;
  int nb = blockIdx.y * 32;
  const float* ip = in + (size_t)l * DM * DM;
  _Float16*    op = out + (size_t)l * DM * DM;
#pragma unroll
  for (int r = threadIdx.y; r < 32; r += 8)
    tile[r][threadIdx.x] = (_Float16)ip[(size_t)(kb + r) * DM + nb + threadIdx.x];
  __syncthreads();
#pragma unroll
  for (int r = threadIdx.y; r < 32; r += 8)
    op[(size_t)(nb + r) * DM + kb + threadIdx.x] = tile[threadIdx.x][r];
}

// ---------------------------------------------------------------------------
// Embedding: x = agent @ emb_W + emb_b + pe. grid = SB, block = 256 (f4 lanes)
// ---------------------------------------------------------------------------
__global__ __launch_bounds__(256) void embed_kernel(
    const float* __restrict__ src, const float* __restrict__ W,
    const float* __restrict__ bias, const float* __restrict__ pe,
    float* __restrict__ x, int agent) {
  int row = blockIdx.x;
  int d4  = threadIdx.x;                        // 256 * 4 = 1024
  int s   = row / BATCH;
  const float* a = src + (size_t)row * (3 * INAG) + agent * INAG;
  f4 acc = ((const f4*)bias)[d4] + ((const f4*)(pe + (size_t)s * DM))[d4];
#pragma unroll
  for (int i = 0; i < INAG; ++i)
    acc += a[i] * ((const f4*)(W + (size_t)i * DM))[d4];
  ((f4*)(x + (size_t)row * DM))[d4] = acc;
}

// ---------------------------------------------------------------------------
// Batch-axis diff (reference diffs along axis=1 == batch):
// ---------------------------------------------------------------------------
__global__ __launch_bounds__(256) void diff_kernel(
    const float* __restrict__ x, float* __restrict__ xv, float* __restrict__ xa) {
  size_t i4x = (size_t)blockIdx.x * 256 + threadIdx.x;   // f4 index
  size_t base = i4x * 4;
  if (base >= SBD) return;
  int b = (int)((base / DM) % BATCH);
  f4 zero = {0.f, 0.f, 0.f, 0.f};
  f4 x0 = ((const f4*)x)[i4x];
  f4 x1 = (b < BATCH - 1) ? ((const f4*)x)[i4x + DM / 4] : zero;
  f4 x2 = (b < BATCH - 2) ? ((const f4*)x)[i4x + DM / 2] : zero;
  f4 v0 = (b < BATCH - 1) ? (x1 - x0) : zero;
  f4 v1 = (b < BATCH - 2) ? (x2 - x1) : zero;
  ((f4*)xv)[i4x] = v0;
  ((f4*)xa)[i4x] = (b < BATCH - 1) ? (v1 - v0) : zero;
}

// ---------------------------------------------------------------------------
// x += xv + xa
// ---------------------------------------------------------------------------
__global__ __launch_bounds__(256) void add3_kernel(
    float* __restrict__ x, const float* __restrict__ xv, const float* __restrict__ xa) {
  size_t i4x = (size_t)blockIdx.x * 256 + threadIdx.x;
  if (i4x * 4 >= SBD) return;
  ((f4*)x)[i4x] = ((const f4*)x)[i4x] + ((const f4*)xv)[i4x] + ((const f4*)xa)[i4x];
}

// ---------------------------------------------------------------------------
// LayerNorm over last dim (1024). One block (256 thr, f4 each) per row.
// ---------------------------------------------------------------------------
__global__ __launch_bounds__(256) void ln_kernel(
    const float* __restrict__ in, const float* __restrict__ g,
    const float* __restrict__ b, float* __restrict__ out) {
  int row = blockIdx.x;
  int tid = threadIdx.x;
  __shared__ float s1[256], s2[256];
  const f4* p = (const f4*)(in + (size_t)row * DM);
  f4 v = p[tid];
  float a  = v[0] + v[1] + v[2] + v[3];
  float sq = v[0] * v[0] + v[1] * v[1] + v[2] * v[2] + v[3] * v[3];
  s1[tid] = a; s2[tid] = sq;
  __syncthreads();
  for (int off = 128; off > 0; off >>= 1) {
    if (tid < off) { s1[tid] += s1[tid + off]; s2[tid] += s2[tid + off]; }
    __syncthreads();
  }
  float mu  = s1[0] * (1.f / DM);
  float var = s2[0] * (1.f / DM) - mu * mu;
  float inv = rsqrtf(var + 1e-5f);
  f4 gv = ((const f4*)g)[tid];
  f4 bv = ((const f4*)b)[tid];
  ((f4*)(out + (size_t)row * DM))[tid] = (v - mu) * inv * gv + bv;
}

// ---------------------------------------------------------------------------
// WMMA GEMM: C[M x 1024] = A(f32) @ Bt(f16, pre-transposed [n][k])
//            + bias (+ residual / + relu per EPI template)
// 256 threads = 8 waves; 128x128 tile; double-buffered LDS; 1 barrier/K-step.
// EPI: 0 = bias, 1 = bias+residual, 2 = bias+relu
// ---------------------------------------------------------------------------
template <int EPI>
__global__ __launch_bounds__(256) void gemm_kernel(
    const float* __restrict__ A, const _Float16* __restrict__ Bt,
    const float* __restrict__ bias, const float* __restrict__ residual,
    float* __restrict__ C) {
  __shared__ __align__(16) _Float16 As[2][TM * TK];  // [m][k]
  __shared__ __align__(16) _Float16 Bs[2][TN * TK];  // [n][k]

  int tid  = threadIdx.x;
  int m0   = blockIdx.x * TM;
  int n0   = blockIdx.y * TN;
  int wave = tid >> 5;
  int lane = tid & 31;
  int wm   = (wave >> 1) * 32;   // 4 waves along M
  int wn   = (wave & 1) * 64;    // 2 waves along N
  int g    = lane >> 4;
  int rl   = lane & 15;

  // Per-thread staging coords: 512 groups of 8 elements; 2 groups per thread.
  int sm[2], sko[2];
#pragma unroll
  for (int j = 0; j < 2; ++j) {
    int lin8 = tid + 256 * j;
    sm[j]  = lin8 >> 2;          // row (m for A, n for B), 0..127
    sko[j] = (lin8 & 3) * 8;     // k offset within tile
  }

  f4 ar[2][2];
  i4 br[2];
  auto loadTile = [&](int k0) {
#pragma unroll
    for (int j = 0; j < 2; ++j) {
      const f4* pa = (const f4*)(A + (size_t)(m0 + sm[j]) * DM + k0 + sko[j]);
      ar[j][0] = pa[0];
      ar[j][1] = pa[1];
      br[j] = *(const i4*)(Bt + (size_t)(n0 + sm[j]) * DM + k0 + sko[j]);
    }
  };
  auto storeTile = [&](int buf) {
#pragma unroll
    for (int j = 0; j < 2; ++j) {
      *(h8*)&As[buf][sm[j] * TK + sko[j]] = pack8(ar[j][0], ar[j][1]);
      *(i4*)&Bs[buf][sm[j] * TK + sko[j]] = br[j];
    }
  };

  v8f zero = {0.f, 0.f, 0.f, 0.f, 0.f, 0.f, 0.f, 0.f};
  v8f acc[2][4];
#pragma unroll
  for (int mi = 0; mi < 2; ++mi)
#pragma unroll
    for (int ni = 0; ni < 4; ++ni) acc[mi][ni] = zero;

  loadTile(0);
  for (int kt = 0; kt < NK; ++kt) {
    int buf = kt & 1;
    storeTile(buf);
    __syncthreads();
    if (kt + 1 < NK) loadTile((kt + 1) * TK);   // overlap with WMMA below

    F16Frag af[2], bf[4];
#pragma unroll
    for (int mi = 0; mi < 2; ++mi) {
      const _Float16* base = &As[buf][(wm + mi * 16 + rl) * TK];
      af[mi].q[0] = *(const i4*)(base + 8 * g);        // k = 8g..8g+7
      af[mi].q[1] = *(const i4*)(base + 16 + 8 * g);   // k = 16+8g..16+8g+7
    }
#pragma unroll
    for (int ni = 0; ni < 4; ++ni) {
      const _Float16* base = &Bs[buf][(wn + ni * 16 + rl) * TK];
      bf[ni].q[0] = *(const i4*)(base + 8 * g);        // k = 8g..8g+7
      bf[ni].q[1] = *(const i4*)(base + 16 + 8 * g);   // k = 16+8g..16+8g+7
    }
#pragma unroll
    for (int mi = 0; mi < 2; ++mi)
#pragma unroll
      for (int ni = 0; ni < 4; ++ni)
        acc[mi][ni] = __builtin_amdgcn_wmma_f32_16x16x32_f16(
            false, af[mi].v, false, bf[ni].v, (short)0, acc[mi][ni], false, false);
  }

  // Epilogue: D layout — lane n = rl, element r -> m = r + 8g
#pragma unroll
  for (int mi = 0; mi < 2; ++mi)
#pragma unroll
    for (int ni = 0; ni < 4; ++ni) {
      int n = n0 + wn + ni * 16 + rl;
      float bn = bias[n];
#pragma unroll
      for (int r = 0; r < 8; ++r) {
        int m = m0 + wm + mi * 16 + r + 8 * g;
        float v = acc[mi][ni][r] + bn;
        if constexpr (EPI == 1) v += residual[(size_t)m * DM + n];
        if constexpr (EPI == 2) v = fmaxf(v, 0.f);
        C[(size_t)m * DM + n] = v;
      }
    }
}

// ---------------------------------------------------------------------------
// Attention: one block per (b,h). K == V (same projection of same tensor).
// scores = q.k / HD ; softmax over 25 ; out = attn @ k
// ---------------------------------------------------------------------------
__global__ __launch_bounds__(256) void attn_kernel(
    const float* __restrict__ Q, const float* __restrict__ KV,
    float* __restrict__ Out) {
  int b = blockIdx.x >> 4;
  int h = blockIdx.x & 15;
  int tid = threadIdx.x;
  __shared__ __align__(16) float qs[SEQ * HD];
  __shared__ __align__(16) float ks[SEQ * HD];
  __shared__ float sc[SEQ * SEQ];

  // f4 loads: 400 vec4 groups per tensor
  for (int i = tid; i < SEQ * HD / 4; i += 256) {
    int s  = i >> 4;                         // HD/4 = 16 groups per row
    int dv = i & 15;
    size_t off = (((size_t)s * BATCH + b) * DM + h * HD) / 4 + dv;
    ((f4*)qs)[i] = ((const f4*)Q)[off];
    ((f4*)ks)[i] = ((const f4*)KV)[off];
  }
  __syncthreads();

  for (int i = tid; i < SEQ * SEQ; i += 256) {
    int qi = i / SEQ, ki = i % SEQ;
    float dot = 0.f;
#pragma unroll
    for (int d = 0; d < HD; ++d) dot += qs[qi * HD + d] * ks[ki * HD + d];
    sc[i] = dot * (1.f / HD);
  }
  __syncthreads();

  if (tid < SEQ) {
    float mx = -3.4e38f;
#pragma unroll
    for (int k = 0; k < SEQ; ++k) mx = fmaxf(mx, sc[tid * SEQ + k]);
    float sum = 0.f;
#pragma unroll
    for (int k = 0; k < SEQ; ++k) {
      float e = __expf(sc[tid * SEQ + k] - mx);
      sc[tid * SEQ + k] = e;
      sum += e;
    }
    float inv = 1.f / sum;
#pragma unroll
    for (int k = 0; k < SEQ; ++k) sc[tid * SEQ + k] *= inv;
  }
  __syncthreads();

  for (int i = tid; i < SEQ * HD / 4; i += 256) {
    int qi = i >> 4;
    int dv = i & 15;
    f4 o = {0.f, 0.f, 0.f, 0.f};
#pragma unroll
    for (int k = 0; k < SEQ; ++k) o += sc[qi * SEQ + k] * ((const f4*)ks)[k * 16 + dv];
    size_t off = (((size_t)qi * BATCH + b) * DM + h * HD) / 4 + dv;
    ((f4*)Out)[off] = o;
  }
}

// ---------------------------------------------------------------------------
// Host orchestration
// ---------------------------------------------------------------------------
extern "C" void kernel_launch(void* const* d_in, const int* in_sizes, int n_in,
                              void* d_out, int out_size, void* d_ws, size_t ws_size,
                              hipStream_t stream) {
  (void)in_sizes; (void)n_in; (void)out_size; (void)ws_size;

  const float* src   = (const float*)d_in[0];
  const float* embW  = (const float*)d_in[1];
  const float* embB  = (const float*)d_in[2];
  const float* pe    = (const float*)d_in[3];
  const float* qW    = (const float*)d_in[4];
  const float* qb    = (const float*)d_in[5];
  const float* oW    = (const float*)d_in[6];
  const float* ob    = (const float*)d_in[7];
  const float* n1g   = (const float*)d_in[8];
  const float* n1b   = (const float*)d_in[9];
  const float* n2g   = (const float*)d_in[10];
  const float* n2b   = (const float*)d_in[11];
  const float* f1W   = (const float*)d_in[12];
  const float* f1b   = (const float*)d_in[13];
  const float* f2W   = (const float*)d_in[14];
  const float* f2b   = (const float*)d_in[15];
  float* out = (float*)d_out;

  // Workspace carve-up
  auto aln = [](size_t x) { return (x + 255) & ~(size_t)255; };
  char* w = (char*)d_ws;
  size_t wsz = aln((size_t)NL * DM * DM * sizeof(_Float16));
  _Float16* qWh = (_Float16*)w;            w += wsz;
  _Float16* oWh = (_Float16*)w;            w += wsz;
  _Float16* f1h = (_Float16*)w;            w += wsz;
  _Float16* f2h = (_Float16*)w;            w += wsz;
  size_t bsz = aln(SBD * sizeof(float));
  float* x  = (float*)w; w += bsz;
  float* xv = (float*)w; w += bsz;
  float* xa = (float*)w; w += bsz;
  float* t0 = (float*)w; w += bsz;
  float* t1 = (float*)w; w += bsz;
  float* t2 = (float*)w; w += bsz;

  dim3 gCvt(DM / 32, DM / 32, NL);
  dim3 bCvt(32, 8);
  convert_transpose_kernel<<<gCvt, bCvt, 0, stream>>>(qW, qWh);
  convert_transpose_kernel<<<gCvt, bCvt, 0, stream>>>(oW, oWh);
  convert_transpose_kernel<<<gCvt, bCvt, 0, stream>>>(f1W, f1h);
  convert_transpose_kernel<<<gCvt, bCvt, 0, stream>>>(f2W, f2h);

  dim3 gGemm(SB / TM, DM / TN);                        // (100, 8)
  dim3 gElem4((unsigned)(SBD / 4 / 256));              // 12800
  dim3 gAttn(BATCH * NH);                              // 8192
  dim3 gEmb(SB);

  for (int a = 0; a < 3; ++a) {
    embed_kernel<<<gEmb, 256, 0, stream>>>(src, embW, embB, pe, x, a);
    for (int l = 0; l < NL; ++l) {
      const _Float16* qWl = qWh + (size_t)l * DM * DM;
      const _Float16* oWl = oWh + (size_t)l * DM * DM;
      const _Float16* f1l = f1h + (size_t)l * DM * DM;
      const _Float16* f2l = f2h + (size_t)l * DM * DM;
      const float* qbl  = qb  + (size_t)l * DM;
      const float* obl  = ob  + (size_t)l * DM;
      const float* f1bl = f1b + (size_t)l * DM;
      const float* f2bl = f2b + (size_t)l * DM;
      const float* n1gl = n1g + (size_t)l * DM;
      const float* n1bl = n1b + (size_t)l * DM;
      const float* n2gl = n2g + (size_t)l * DM;
      const float* n2bl = n2b + (size_t)l * DM;

      // velocity / acceleration diffs, then pre-norms (n1)
      diff_kernel<<<gElem4, 256, 0, stream>>>(x, xv, xa);
      ln_kernel<<<SB, 256, 0, stream>>>(x,  n1gl, n1bl, x);
      ln_kernel<<<SB, 256, 0, stream>>>(xv, n1gl, n1bl, xv);
      ln_kernel<<<SB, 256, 0, stream>>>(xa, n1gl, n1bl, xa);

      // MHA 1: q=k=v = proj(xn); x = xn + attn@oW+ob
      gemm_kernel<0><<<gGemm, 256, 0, stream>>>(x, qWl, qbl, nullptr, t0);
      attn_kernel<<<gAttn, 256, 0, stream>>>(t0, t0, t1);
      gemm_kernel<1><<<gGemm, 256, 0, stream>>>(t1, oWl, obl, x, x);

      // Shared Q projection of updated x for MHA 2 & 3
      gemm_kernel<0><<<gGemm, 256, 0, stream>>>(x, qWl, qbl, nullptr, t1);

      // MHA 2 (velocity): kv = proj(vn); xv = vn + attn@oW+ob
      gemm_kernel<0><<<gGemm, 256, 0, stream>>>(xv, qWl, qbl, nullptr, t0);
      attn_kernel<<<gAttn, 256, 0, stream>>>(t1, t0, t2);
      gemm_kernel<1><<<gGemm, 256, 0, stream>>>(t2, oWl, obl, xv, xv);

      // MHA 3 (acceleration)
      gemm_kernel<0><<<gGemm, 256, 0, stream>>>(xa, qWl, qbl, nullptr, t0);
      attn_kernel<<<gAttn, 256, 0, stream>>>(t1, t0, t2);
      gemm_kernel<1><<<gGemm, 256, 0, stream>>>(t2, oWl, obl, xa, xa);

      // Combine streams, norm (n2), FFN with fused relu + residual
      add3_kernel<<<gElem4, 256, 0, stream>>>(x, xv, xa);
      ln_kernel<<<SB, 256, 0, stream>>>(x, n2gl, n2bl, x);
      gemm_kernel<2><<<gGemm, 256, 0, stream>>>(x, f1l, f1bl, nullptr, t0);
      float* dst = (l == NL - 1) ? (out + (size_t)a * SBD) : x;
      gemm_kernel<1><<<gGemm, 256, 0, stream>>>(t0, f2l, f2bl, x, dst);
    }
  }
}
